// ManyToOne_GRU_34943853920861
// MI455X (gfx1250) — matrix operand order repaired and checked by reading, not compiled, not run on hardware
//
#include <hip/hip_runtime.h>
#include <hip/hip_bf16.h>

// ---------------------------------------------------------------------------
// ManyToOne GRU for MI455X (gfx1250, wave32, WMMA bf16 + async-LDS path)
//   N=256 batch, L=512 steps, I=512 input, H=1024 hidden
// ---------------------------------------------------------------------------

typedef __bf16 bf16;
typedef __attribute__((ext_vector_type(16))) __bf16 v16bf;
typedef __attribute__((ext_vector_type(8)))  __bf16 v8bf;
typedef __attribute__((ext_vector_type(8)))  float  v8f;

constexpr int NB = 256;   // batch
constexpr int LT = 512;   // sequence length
constexpr int ID = 512;   // input size
constexpr int HD = 1024;  // hidden size
constexpr int H3 = 3 * HD;

union AFrag { v16bf v; v8bf h[2]; };

__device__ __forceinline__ float fast_sigmoid(float x) {
    return 1.0f / (1.0f + __expf(-x));
}
__device__ __forceinline__ float fast_tanh(float x) {
    float ax = fabsf(x);
    float e  = __expf(-2.0f * ax);
    float r  = (1.0f - e) / (1.0f + e);
    return copysignf(r, x);
}

__device__ __forceinline__ unsigned pack2bf(float a, float b) {
    union { __bf16 h[2]; unsigned u; } p;
    p.h[0] = (__bf16)a;
    p.h[1] = (__bf16)b;
    return p.u;
}

// CDNA5 async copy: 16B global -> LDS, tracked on ASYNCcnt (no VGPR data path)
__device__ __forceinline__ void async_b128_to_lds(unsigned lds_off,
                                                  const void* gptr) {
    asm volatile("global_load_async_to_lds_b128 %0, %1, off"
                 :: "v"(lds_off), "v"((unsigned long long)(size_t)gptr)
                 : "memory");
}
__device__ __forceinline__ void wait_async0() {
    asm volatile("s_wait_asynccnt 0x0" ::: "memory");
}

// f32 -> bf16 elementwise convert
__global__ void cvt_f32_bf16_kernel(const float* __restrict__ src,
                                    bf16* __restrict__ dst, int n) {
    int i = blockIdx.x * blockDim.x + threadIdx.x;
    if (i < n) dst[i] = (bf16)src[i];
}

// init h: copy f32 and make bf16 shadow
__global__ void init_h_kernel(const float* __restrict__ src,
                              float* __restrict__ dst_f,
                              bf16* __restrict__ dst_b, int n) {
    int i = blockIdx.x * blockDim.x + threadIdx.x;
    if (i < n) {
        float v = src[i];
        dst_f[i] = v;
        dst_b[i] = (bf16)v;
    }
}

__global__ void copy_f32_kernel(const float* __restrict__ src,
                                float* __restrict__ dst, int n) {
    int i = blockIdx.x * blockDim.x + threadIdx.x;
    if (i < n) dst[i] = src[i];
}

// ---------------------------------------------------------------------------
// One GRU time step, fully fused:
//   gi = xt @ W_ih^T   (K = ID)
//   gh = h  @ W_hh^T   (K = HD)
//   r = sig(gi_r+gh_r+b), z = sig(...), n = tanh(gi_n + b + r*(gh_n + b))
//   h' = (1-z)*n + z*h
// Grid: (H/128, N/16).  Block: 32x8 (8 waves). Wave w owns 16 h-columns.
// ---------------------------------------------------------------------------
__global__ __launch_bounds__(256) void gru_step_kernel(
    const float* __restrict__ x,     // [N, L, I] f32
    const bf16*  __restrict__ Wih,   // [3H, I] bf16
    const bf16*  __restrict__ Whh,   // [3H, H] bf16
    const float* __restrict__ b_ih,  // [3H]
    const float* __restrict__ b_hh,  // [3H]
    const float* __restrict__ h_in_f,// [N, H] f32
    const bf16*  __restrict__ h_in_b,// [N, H] bf16
    float* __restrict__ h_out_f,
    bf16*  __restrict__ h_out_b,
    int t)
{
    __shared__ bf16 xs[16 * ID];   // 16 KB: x tile, converted to bf16
    __shared__ bf16 hs[16 * HD];   // 32 KB: h tile (bf16)

    const int lane = threadIdx.x;            // 0..31
    const int wave = threadIdx.y;            // 0..7
    const int tid  = wave * 32 + lane;       // 0..255
    const int n0   = blockIdx.y * 16;        // batch-row tile base
    const int c0   = blockIdx.x * 128 + wave * 16;  // this wave's h-col base

    // ---- stage h tile via async global->LDS (rows n0..n0+15 contiguous) ---
    {
        const char* src = (const char*)(h_in_b + (size_t)n0 * HD);  // 32 KB
        for (int c = tid; c < 2048; c += 256) {                      // 16B each
            unsigned lds_off = (unsigned)(size_t)((char*)hs + c * 16);
            async_b128_to_lds(lds_off, src + c * 16);
        }
    }
    // ---- stage x tile: rows n0..n0+15 of x[:, t, :], f32 -> bf16 ----------
    {
        unsigned* xd = (unsigned*)xs;
        for (int idx = tid; idx < 16 * ID / 2; idx += 256) {  // pairs
            int r = idx >> 8;            // 256 pairs per row
            int c = idx & 255;
            const float2 v = ((const float2*)(x + ((size_t)(n0 + r) * LT + t) * ID))[c];
            xd[idx] = pack2bf(v.x, v.y);
        }
    }
    wait_async0();
    __syncthreads();

    const int half = lane >> 4;      // K-half select (A & B fragments)
    const int bn   = lane & 15;      // B fragment: column within 16-col slice
    const int col  = c0 + bn;        // global h column this lane's B covers

    v8f accR  = {};   // i_r + h_r
    v8f accZ  = {};   // i_z + h_z
    v8f accIN = {};   // i_n
    v8f accHN = {};   // h_n

    // ---- phase 1: gi += xt @ W_ih^T over K = ID ---------------------------
    {
        const bf16* pr = Wih + (size_t)(0 * HD + col) * ID + 16 * half;
        const bf16* pz = Wih + (size_t)(1 * HD + col) * ID + 16 * half;
        const bf16* pn = Wih + (size_t)(2 * HD + col) * ID + 16 * half;
        __builtin_prefetch(pr, 0, 1);
        __builtin_prefetch(pz, 0, 1);
        __builtin_prefetch(pn, 0, 1);
        const bf16* arow = &xs[(size_t)(lane & 15) * ID + 16 * half];
#pragma unroll 4
        for (int k0 = 0; k0 < ID; k0 += 32) {
            AFrag a;
            a.h[0] = *(const v8bf*)(arow + k0);
            a.h[1] = *(const v8bf*)(arow + k0 + 8);
            v16bf br = *(const v16bf*)(pr + k0);
            v16bf bz = *(const v16bf*)(pz + k0);
            v16bf bb = *(const v16bf*)(pn + k0);
            accR  = __builtin_amdgcn_wmma_f32_16x16x32_bf16(
                        false, a.v, false, br, (short)0, accR,  false, false);
            accZ  = __builtin_amdgcn_wmma_f32_16x16x32_bf16(
                        false, a.v, false, bz, (short)0, accZ,  false, false);
            accIN = __builtin_amdgcn_wmma_f32_16x16x32_bf16(
                        false, a.v, false, bb, (short)0, accIN, false, false);
        }
    }

    // ---- phase 2: gh += h @ W_hh^T over K = HD ----------------------------
    {
        const bf16* qr = Whh + (size_t)(0 * HD + col) * HD + 16 * half;
        const bf16* qz = Whh + (size_t)(1 * HD + col) * HD + 16 * half;
        const bf16* qn = Whh + (size_t)(2 * HD + col) * HD + 16 * half;
        __builtin_prefetch(qr, 0, 1);
        __builtin_prefetch(qz, 0, 1);
        __builtin_prefetch(qn, 0, 1);
        const bf16* arow = &hs[(size_t)(lane & 15) * HD + 16 * half];
#pragma unroll 4
        for (int k0 = 0; k0 < HD; k0 += 32) {
            AFrag a;
            a.h[0] = *(const v8bf*)(arow + k0);
            a.h[1] = *(const v8bf*)(arow + k0 + 8);
            v16bf br = *(const v16bf*)(qr + k0);
            v16bf bz = *(const v16bf*)(qz + k0);
            v16bf bb = *(const v16bf*)(qn + k0);
            accR  = __builtin_amdgcn_wmma_f32_16x16x32_bf16(
                        false, a.v, false, br, (short)0, accR,  false, false);
            accZ  = __builtin_amdgcn_wmma_f32_16x16x32_bf16(
                        false, a.v, false, bz, (short)0, accZ,  false, false);
            accHN = __builtin_amdgcn_wmma_f32_16x16x32_bf16(
                        false, a.v, false, bb, (short)0, accHN, false, false);
        }
    }

    // ---- gates + state update --------------------------------------------
    // C/D layout: VGPR j, lanes 0-15 -> row j (N=lane); lanes 16-31 -> row j+8
    const float bihr = b_ih[col], bihz = b_ih[HD + col], bihn = b_ih[2 * HD + col];
    const float bhhr = b_hh[col], bhhz = b_hh[HD + col], bhhn = b_hh[2 * HD + col];
    const int rbase = n0 + half * 8;
#pragma unroll
    for (int j = 0; j < 8; ++j) {
        const size_t off = (size_t)(rbase + j) * HD + col;
        float r  = fast_sigmoid(accR[j] + bihr + bhhr);
        float z  = fast_sigmoid(accZ[j] + bihz + bhhz);
        float nn = fast_tanh(accIN[j] + bihn + r * (accHN[j] + bhhn));
        float ho = h_in_f[off];
        float hn = (1.0f - z) * nn + z * ho;
        h_out_f[off] = hn;
        h_out_b[off] = (bf16)hn;
    }
}

// ---------------------------------------------------------------------------
// Head: y = h_final @ W_out^T + b_out   (N x I, K = HD)
// Grid: (I/128, N/16), block 32x8.
// ---------------------------------------------------------------------------
__global__ __launch_bounds__(256) void gru_head_kernel(
    const bf16*  __restrict__ hb,    // [N, H] bf16 (final hidden)
    const bf16*  __restrict__ Wout,  // [I, H] bf16
    const float* __restrict__ b_out, // [I]
    float* __restrict__ y)           // [N, I]
{
    __shared__ bf16 hs[16 * HD];     // 32 KB

    const int lane = threadIdx.x;
    const int wave = threadIdx.y;
    const int tid  = wave * 32 + lane;
    const int n0   = blockIdx.y * 16;
    const int c0   = blockIdx.x * 128 + wave * 16;

    {
        const char* src = (const char*)(hb + (size_t)n0 * HD);   // 32 KB block
        for (int c = tid; c < 2048; c += 256) {
            unsigned lds_off = (unsigned)(size_t)((char*)hs + c * 16);
            async_b128_to_lds(lds_off, src + c * 16);
        }
    }
    wait_async0();
    __syncthreads();

    const int half = lane >> 4;
    const int bn   = lane & 15;
    const int col  = c0 + bn;

    v8f acc = {};
    const bf16* p    = Wout + (size_t)col * HD + 16 * half;
    __builtin_prefetch(p, 0, 1);
    const bf16* arow = &hs[(size_t)(lane & 15) * HD + 16 * half];
#pragma unroll 4
    for (int k0 = 0; k0 < HD; k0 += 32) {
        AFrag a;
        a.h[0] = *(const v8bf*)(arow + k0);
        a.h[1] = *(const v8bf*)(arow + k0 + 8);
        v16bf b = *(const v16bf*)(p + k0);
        acc = __builtin_amdgcn_wmma_f32_16x16x32_bf16(
                  false, a.v, false, b, (short)0, acc, false, false);
    }

    const float bo = b_out[col];
    const int rbase = n0 + half * 8;
#pragma unroll
    for (int j = 0; j < 8; ++j) {
        y[(size_t)(rbase + j) * ID + col] = acc[j] + bo;
    }
}

// ---------------------------------------------------------------------------
extern "C" void kernel_launch(void* const* d_in, const int* in_sizes, int n_in,
                              void* d_out, int out_size, void* d_ws, size_t ws_size,
                              hipStream_t stream) {
    const float* x     = (const float*)d_in[0];   // [N, L, I]
    const float* h0    = (const float*)d_in[1];   // [N, H]
    const float* W_ih  = (const float*)d_in[2];   // [3H, I]
    const float* b_ih  = (const float*)d_in[3];   // [3H]
    const float* W_hh  = (const float*)d_in[4];   // [3H, H]
    const float* b_hh  = (const float*)d_in[5];   // [3H]
    const float* W_out = (const float*)d_in[6];   // [I, H]
    const float* b_out = (const float*)d_in[7];   // [I]

    // workspace layout (bytes)
    char* base = (char*)d_ws;
    size_t oWih  = 0;                              // 3H*I  bf16 = 3,145,728
    size_t oWhh  = oWih  + (size_t)H3 * ID * 2;    // 3H*H  bf16 = 6,291,456
    size_t oWout = oWhh  + (size_t)H3 * HD * 2;    // I*H   bf16 = 1,048,576
    size_t oHf0  = oWout + (size_t)ID * HD * 2;    // N*H   f32
    size_t oHf1  = oHf0  + (size_t)NB * HD * 4;
    size_t oHb0  = oHf1  + (size_t)NB * HD * 4;    // N*H   bf16
    size_t oHb1  = oHb0  + (size_t)NB * HD * 2;
    size_t total = oHb1  + (size_t)NB * HD * 2;
    if (ws_size < total) return;

    bf16*  Wih_b  = (bf16*)(base + oWih);
    bf16*  Whh_b  = (bf16*)(base + oWhh);
    bf16*  Wout_b = (bf16*)(base + oWout);
    float* hf[2]  = { (float*)(base + oHf0), (float*)(base + oHf1) };
    bf16*  hb[2]  = { (bf16*)(base + oHb0), (bf16*)(base + oHb1) };

    // ---- prep: weight converts + h init ----------------------------------
    {
        int n = H3 * ID;
        cvt_f32_bf16_kernel<<<(n + 255) / 256, 256, 0, stream>>>(W_ih, Wih_b, n);
        n = H3 * HD;
        cvt_f32_bf16_kernel<<<(n + 255) / 256, 256, 0, stream>>>(W_hh, Whh_b, n);
        n = ID * HD;
        cvt_f32_bf16_kernel<<<(n + 255) / 256, 256, 0, stream>>>(W_out, Wout_b, n);
        n = NB * HD;
        init_h_kernel<<<(n + 255) / 256, 256, 0, stream>>>(h0, hf[0], hb[0], n);
    }

    // ---- recurrence: 512 dependent step kernels (ping-pong h) ------------
    dim3 stepGrid(HD / 128, NB / 16);   // (8, 16)
    dim3 blk(32, 8);
    int cur = 0;
    for (int t = 0; t < LT; ++t) {
        gru_step_kernel<<<stepGrid, blk, 0, stream>>>(
            x, Wih_b, Whh_b, b_ih, b_hh,
            hf[cur], hb[cur], hf[1 - cur], hb[1 - cur], t);
        cur ^= 1;
    }

    // ---- head + pack outputs: d_out = [ y (N*I) | h_final (N*H) ] --------
    float* y_out = (float*)d_out;
    float* h_out = (float*)d_out + (size_t)NB * ID;
    dim3 headGrid(ID / 128, NB / 16);   // (4, 16)
    gru_head_kernel<<<headGrid, blk, 0, stream>>>(hb[cur], Wout_b, b_out, y_out);
    {
        int n = NB * HD;
        copy_f32_kernel<<<(n + 255) / 256, 256, 0, stream>>>(hf[cur], h_out, n);
    }
}
